// ModernHopfieldMemory_88252987998469
// MI455X (gfx1250) — compile-verified, matrix operand order checked by reading, not run on hardware
//
#include <hip/hip_runtime.h>

// ---------- types ----------
typedef __attribute__((ext_vector_type(16))) __bf16 v16bf;
typedef __attribute__((ext_vector_type(8)))  float  v8f;
typedef __attribute__((ext_vector_type(4)))  unsigned int u32x4;

union FragU { u32x4 q[2]; v16bf v; };

__device__ __forceinline__ unsigned short f2bf(float f) {
    unsigned int u = __float_as_uint(f);
    u += 0x7FFFu + ((u >> 16) & 1u);          // round-to-nearest-even
    return (unsigned short)(u >> 16);
}
__device__ __forceinline__ float bf2f(unsigned short h) {
    return __uint_as_float(((unsigned int)h) << 16);
}

// Load one 16x32 bf16 WMMA fragment for a lane.
// base points at rowStart + k0 + (lane>>4)*8 elements; 16B aligned.
// ISA layout: lanes 0-15 hold K {0..7,16..23}, lanes 16-31 hold K {8..15,24..31}.
__device__ __forceinline__ v16bf load_frag(const unsigned short* base) {
    const u32x4* p = reinterpret_cast<const u32x4*>(base);
    FragU u;
    u.q[0] = p[0];   // K  {+0..+7}
    u.q[1] = p[2];   // K {+16..+23} (offset 16 elements = 32 bytes)
    return u.v;
}

// ---------- GEMM: D = A (MxK, row-major bf16) x B^T (B is NxK, row-major bf16) ----------
// Wave tile 64x64 (4x4 WMMA subtiles): 16 WMMAs per 16 b128 loads (1.0 load/WMMA).
// Workgroup: 8 waves as 2(M) x 4(N) -> 128 x 256 output tile.
// mode 0: outB[idx]   = bf16(acc)                              (k, sim)
// mode 1: outF = acc; outB = bf16(acc)                          (q -> state init)
// mode 2: s = 0.5*mixSrc + 0.5*acc; outF = s; outB = bf16(s)    (state update)
// mode 3: outF = acc                                            (final output)
__global__ __launch_bounds__(256)
void gemm_abT_bf16(const unsigned short* __restrict__ A,
                   const unsigned short* __restrict__ B,
                   int M, int N, int K,
                   float* __restrict__ outF,
                   unsigned short* __restrict__ outB,
                   const float* __restrict__ mixSrc,
                   int mode)
{
    const int lane = threadIdx.x & 31;
    const int wave = threadIdx.x >> 5;      // 0..7
    const int l16  = lane & 15;
    const int lh   = lane >> 4;             // 0/1

    const int waveM = wave & 1;             // 2 waves along M
    const int waveN = wave >> 1;            // 4 waves along N
    const int tileM0 = blockIdx.y * 128 + waveM * 64;   // 64 rows per wave
    const int tileN0 = blockIdx.x * 256 + waveN * 64;   // 64 cols per wave

    v8f acc[4][4];
    #pragma unroll
    for (int i = 0; i < 4; ++i)
        #pragma unroll
        for (int j = 0; j < 4; ++j)
            #pragma unroll
            for (int e = 0; e < 8; ++e) acc[i][j][e] = 0.0f;

    const unsigned short* aRow[4];
    const unsigned short* bRow[4];
    #pragma unroll
    for (int i = 0; i < 4; ++i) {
        aRow[i] = A + (size_t)(tileM0 + i * 16 + l16) * K;
        bRow[i] = B + (size_t)(tileN0 + i * 16 + l16) * K;
    }

    for (int k0 = 0; k0 < K; k0 += 32) {
        const int ko = k0 + lh * 8;
        v16bf a[4], b[4];
        #pragma unroll
        for (int i = 0; i < 4; ++i) a[i] = load_frag(aRow[i] + ko);
        #pragma unroll
        for (int j = 0; j < 4; ++j) b[j] = load_frag(bRow[j] + ko);

        #pragma unroll
        for (int i = 0; i < 4; ++i)
            #pragma unroll
            for (int j = 0; j < 4; ++j)
                acc[i][j] = __builtin_amdgcn_wmma_f32_16x16x32_bf16(
                    false, a[i], false, b[j], (short)0, acc[i][j], false, false);
    }

    // C/D layout: VGPR e, lanes0-15 -> M=e, N=lane; lanes16-31 -> M=e+8, N=lane-16
    #pragma unroll
    for (int i = 0; i < 4; ++i) {
        #pragma unroll
        for (int j = 0; j < 4; ++j) {
            #pragma unroll
            for (int e = 0; e < 8; ++e) {
                const int row = tileM0 + i * 16 + e + lh * 8;
                const int col = tileN0 + j * 16 + l16;
                const size_t idx = (size_t)row * N + col;
                const float v = acc[i][j][e];
                if (mode == 0) {
                    outB[idx] = f2bf(v);
                } else if (mode == 1) {
                    outF[idx] = v;
                    outB[idx] = f2bf(v);
                } else if (mode == 2) {
                    const float s = 0.5f * mixSrc[idx] + 0.5f * v;
                    outF[idx] = s;
                    outB[idx] = f2bf(s);
                } else {
                    outF[idx] = v;
                }
            }
        }
    }
}

// ---------- row softmax over bf16 sim -> bf16 attn ----------
__global__ __launch_bounds__(256)
void softmax_rows_bf16(const unsigned short* __restrict__ sim,
                       unsigned short* __restrict__ attn, int cols)
{
    __shared__ float red[256];
    const int row = blockIdx.x;
    const int tid = threadIdx.x;
    const unsigned short* srow = sim + (size_t)row * cols;

    float m = -3.4e38f;
    for (int c = tid; c < cols; c += 256) m = fmaxf(m, bf2f(srow[c]));
    red[tid] = m; __syncthreads();
    for (int s = 128; s > 0; s >>= 1) { if (tid < s) red[tid] = fmaxf(red[tid], red[tid + s]); __syncthreads(); }
    m = red[0]; __syncthreads();

    float sum = 0.0f;
    for (int c = tid; c < cols; c += 256) sum += __expf(bf2f(srow[c]) - m);
    red[tid] = sum; __syncthreads();
    for (int s = 128; s > 0; s >>= 1) { if (tid < s) red[tid] += red[tid + s]; __syncthreads(); }
    sum = red[0];

    const float inv = 1.0f / sum;
    unsigned short* arow = attn + (size_t)row * cols;
    for (int c = tid; c < cols; c += 256) arow[c] = f2bf(__expf(bf2f(srow[c]) - m) * inv);
}

// ---------- f32 -> bf16 convert ----------
__global__ void cvt_f32_bf16(const float* __restrict__ in, unsigned short* __restrict__ out, size_t n)
{
    size_t i = (size_t)blockIdx.x * blockDim.x + threadIdx.x;
    const size_t stride = (size_t)gridDim.x * blockDim.x;
    for (; i < n; i += stride) out[i] = f2bf(in[i]);
}

// ---------- f32 (rows x cols) -> bf16 transpose (cols x rows) ----------
__global__ __launch_bounds__(256)
void transpose_f32_to_bf16(const float* __restrict__ in, unsigned short* __restrict__ out,
                           int rows, int cols)
{
    __shared__ float tile[32][33];
    const int bx = blockIdx.x * 32;   // col base (input)
    const int by = blockIdx.y * 32;   // row base (input)
    const int tx = threadIdx.x & 31;
    const int ty = threadIdx.x >> 5;  // 0..7
    #pragma unroll
    for (int j = 0; j < 4; ++j) {
        const int r = by + ty + j * 8;
        tile[ty + j * 8][tx] = in[(size_t)r * cols + bx + tx];
    }
    __syncthreads();
    #pragma unroll
    for (int j = 0; j < 4; ++j) {
        const int c = bx + ty + j * 8;
        out[(size_t)c * rows + by + tx] = f2bf(tile[tx][ty + j * 8]);
    }
}

// ---------- orchestration ----------
extern "C" void kernel_launch(void* const* d_in, const int* in_sizes, int n_in,
                              void* d_out, int out_size, void* d_ws, size_t ws_size,
                              hipStream_t stream)
{
    (void)in_sizes; (void)n_in; (void)out_size; (void)ws_size;
    const int C  = 1024;          // feature dim
    const int Mp = 4096;          // number of patterns
    const int R  = 4 * 2048;      // B*T tokens = 8192

    const float* x        = (const float*)d_in[0];
    const float* patterns = (const float*)d_in[1];
    const float* Wq       = (const float*)d_in[2];
    const float* Wk       = (const float*)d_in[3];
    float* out = (float*)d_out;

    char* ws = (char*)d_ws;
    size_t off = 0;
    auto alloc = [&](size_t bytes) -> char* {
        char* p = ws + off;
        off = (off + bytes + 255) & ~(size_t)255;
        return p;
    };
    unsigned short* xb    = (unsigned short*)alloc((size_t)R  * C  * 2);
    unsigned short* Wqb   = (unsigned short*)alloc((size_t)C  * C  * 2);
    unsigned short* Wkb   = (unsigned short*)alloc((size_t)C  * C  * 2);
    unsigned short* patb  = (unsigned short*)alloc((size_t)Mp * C  * 2);
    unsigned short* patTb = (unsigned short*)alloc((size_t)C  * Mp * 2);  // patterns^T (C x M)
    unsigned short* kb    = (unsigned short*)alloc((size_t)Mp * C  * 2);
    unsigned short* stb   = (unsigned short*)alloc((size_t)R  * C  * 2);  // state bf16
    float*          stf   = (float*)         alloc((size_t)R  * C  * 4);  // state f32
    unsigned short* simb  = (unsigned short*)alloc((size_t)R  * Mp * 2);
    unsigned short* attnb = (unsigned short*)alloc((size_t)R  * Mp * 2);

    // precision staging
    cvt_f32_bf16<<<2048, 256, 0, stream>>>(x, xb, (size_t)R * C);
    cvt_f32_bf16<<<1024, 256, 0, stream>>>(Wq, Wqb, (size_t)C * C);
    cvt_f32_bf16<<<1024, 256, 0, stream>>>(Wk, Wkb, (size_t)C * C);
    cvt_f32_bf16<<<2048, 256, 0, stream>>>(patterns, patb, (size_t)Mp * C);
    transpose_f32_to_bf16<<<dim3(C / 32, Mp / 32), 256, 0, stream>>>(patterns, patTb, Mp, C);

    // q = x @ Wq^T -> state (f32 + bf16)
    gemm_abT_bf16<<<dim3(C / 256, R / 128), 256, 0, stream>>>(xb, Wqb, R, C, C, stf, stb, nullptr, 1);
    // k = patterns @ Wk^T -> bf16
    gemm_abT_bf16<<<dim3(C / 256, Mp / 128), 256, 0, stream>>>(patb, Wkb, Mp, C, C, nullptr, kb, nullptr, 0);

    // 3 Hopfield steps + final readout (BETA = 1.0 folded away)
    for (int it = 0; it < 4; ++it) {
        // sim = state @ k^T
        gemm_abT_bf16<<<dim3(Mp / 256, R / 128), 256, 0, stream>>>(stb, kb, R, Mp, C, nullptr, simb, nullptr, 0);
        softmax_rows_bf16<<<R, 256, 0, stream>>>(simb, attnb, Mp);
        if (it < 3) {
            // state = 0.5*state + 0.5*(attn @ patterns) ; patterns via patT^T
            gemm_abT_bf16<<<dim3(C / 256, R / 128), 256, 0, stream>>>(attnb, patTb, R, C, Mp, stf, stb, stf, 2);
        } else {
            // output = attn @ patterns
            gemm_abT_bf16<<<dim3(C / 256, R / 128), 256, 0, stream>>>(attnb, patTb, R, C, Mp, out, nullptr, nullptr, 3);
        }
    }
}
